// GS_9019431322054
// MI455X (gfx1250) — matrix operand order repaired and checked by reading, not compiled, not run on hardware
//
#include <hip/hip_runtime.h>
#include <hip/hip_bf16.h>
#include <math.h>

typedef _Float16 v16h __attribute__((ext_vector_type(16)));
typedef float    v8f  __attribute__((ext_vector_type(8)));

#define NPTS   (256*256*8)
#define HWPIX  (256*256)
#define IMGW   256
#define NSAMP  4

__device__ __forceinline__ float softplusf(float x) {
    return fmaxf(x, 0.f) + log1pf(__expf(-fabsf(x)));
}
__device__ __forceinline__ float sigmoidf(float x) {
    return 1.f / (1.f + __expf(-x));
}
__device__ __forceinline__ void softmax4(float* w) {
    float m = fmaxf(fmaxf(w[0], w[1]), fmaxf(w[2], w[3]));
    float e0 = __expf(w[0] - m), e1 = __expf(w[1] - m);
    float e2 = __expf(w[2] - m), e3 = __expf(w[3] - m);
    float inv = 1.f / (e0 + e1 + e2 + e3);
    w[0] = e0 * inv; w[1] = e1 * inv; w[2] = e2 * inv; w[3] = e3 * inv;
}

// =====================================================================
// Fragment-ordered LDS staging for WMMA (CDNA5 ISA 7.12.2, wave32).
//
// A fragment (16x32 f16): lane L holds row M=L%16; half h maps to
//   K = kt*32 + (h/8)*16 + (L/16)*8 + h%8
// B fragment (32x16 f16): lane L holds col N=L%16; half h maps to
//   K = kt*32 + (h/8)*16 + (L/16)*8 + h%8
// We store staging buffers as [frag][lane][16 halfs] so each lane's
// fragment is a single contiguous 32-byte LDS read (2x ds_load_b128).
// =====================================================================

// =====================================================================
// Kernel 1: per-point stage (agg MLPs + sigma + radiance) with WMMA
//   block = 128 threads = 4 waves; each wave = 4 points = 16 GEMM rows
// =====================================================================
__global__ __launch_bounds__(128)
void k1_point(const float* __restrict__ vox,      // [N][8]
              const float* __restrict__ fin,      // [N][4][23]
              const float* __restrict__ view_w, const float* __restrict__ view_b,
              const float* __restrict__ glob_w, const float* __restrict__ glob_b,
              const float* __restrict__ aggw_w, const float* __restrict__ aggw_b,
              const float* __restrict__ fc_w,   const float* __restrict__ fc_b,
              const float* __restrict__ col1_w, const float* __restrict__ col1_b,
              const float* __restrict__ col2_w, const float* __restrict__ col2_b,
              const float* __restrict__ sig_w,  const float* __restrict__ sig_b,
              float* __restrict__ Xout,          // [N][24]
              float* __restrict__ SIGout,        // [N]
              float* __restrict__ RADout)        // [N][3]
{
    // B weights in fragment order: [kt*2+nt][lane][16]
    __shared__ alignas(32) _Float16 sGlobW[4 * 32 * 16];
    __shared__ alignas(32) _Float16 sCol1W[4 * 32 * 16];
    __shared__ float    sF  [4][16 * 23];
    __shared__ float    sImg[4][16 * 19];
    __shared__ float    sVar[4][4 * 19];
    __shared__ float    sAvg[4][4 * 19];
    // A staging in fragment order: [kt][lane][16]
    __shared__ alignas(32) _Float16 sA[4][2 * 32 * 16];
    __shared__ float    sG  [4][16 * 32];
    __shared__ float    sW4 [4][16];
    __shared__ float    sIMF[4][4 * 32];
    __shared__ float    sXB [4][4 * 24];
    __shared__ float    sH  [4][16 * 32];

    int tid = threadIdx.x;
    // stage B matrices (fragment order), zero-padded K and N
    for (int i = tid; i < 4 * 32 * 16; i += 128) {
        int frag = i >> 9, rem = i & 511, l2 = rem >> 4, h = rem & 15;
        int kt = frag >> 1, nt = frag & 1;
        int col = nt * 16 + (l2 & 15);
        int K   = kt * 32 + ((h >> 3) << 4) + ((l2 >> 4) & 1) * 8 + (h & 7);
        sGlobW[i] = (_Float16)((K < 57) ? glob_w[K * 32 + col] : 0.f);
        sCol1W[i] = (_Float16)((K < 47 && col < 24) ? col1_w[K * 24 + col] : 0.f);
    }
    __syncthreads();

    int wave = tid >> 5, lane = tid & 31;
    int p0 = (blockIdx.x * 4 + wave) * 4;       // first of this wave's 4 points
    float* F = sF[wave];

    // load f rows (global row = p0*4 + r, contiguous)
    for (int i = lane; i < 16 * 23; i += 32) {
        int r = i / 23, c = i - r * 23;
        F[i] = fin[(size_t)(p0 * 4 + r) * 23 + c];
    }

    // view MLP (4->19) + add
    if (lane < 16) {
        int r = lane;
        const float* d4 = &F[r * 23 + 19];
        for (int o = 0; o < 19; ++o) {
            float acc = view_b[o];
            for (int i2 = 0; i2 < 4; ++i2) acc += d4[i2] * view_w[i2 * 19 + o];
            sImg[wave][r * 19 + o] = F[r * 23 + o] + fmaxf(acc, 0.f);
        }
    }
    // per-point mean / var (ddof=1)
    for (int i = lane; i < 4 * 19; i += 32) {
        int p = i / 19, ch = i - p * 19;
        float s = 0.f, s2 = 0.f;
        for (int sm = 0; sm < 4; ++sm) {
            float v = sImg[wave][(p * 4 + sm) * 19 + ch];
            s += v; s2 += v * v;
        }
        float m = s * 0.25f;
        sAvg[wave][i] = m;
        sVar[wave][i] = (s2 - s * m) * (1.f / 3.f);
    }
    // pack feat A-matrix (fragment order): logical [row][K] with
    //   [img(19) | var(19) | avg(19) | 0] K-padded to 64
    _Float16* A = sA[wave];
    for (int i = lane; i < 2 * 32 * 16; i += 32) {
        int kt = i >> 9, rem = i & 511, l2 = rem >> 4, h = rem & 15;
        int row = l2 & 15, p = row >> 2;
        int k = kt * 32 + ((h >> 3) << 4) + ((l2 >> 4) & 1) * 8 + (h & 7);
        float v = 0.f;
        if      (k < 19) v = sImg[wave][row * 19 + k];
        else if (k < 38) v = sVar[wave][p * 19 + (k - 19)];
        else if (k < 57) v = sAvg[wave][p * 19 + (k - 38)];
        A[i] = (_Float16)v;
    }

    // ---- WMMA GEMM #1: [16,57] x [57,32] -> gfeat ----
    {
        const v16h* Af = reinterpret_cast<const v16h*>(A);
        const v16h* Bf = reinterpret_cast<const v16h*>(sGlobW);
        v16h a0 = Af[lane];
        v16h a1 = Af[32 + lane];
#pragma unroll
        for (int nt = 0; nt < 2; ++nt) {
            v16h b0 = Bf[(0 * 2 + nt) * 32 + lane];
            v16h b1 = Bf[(1 * 2 + nt) * 32 + lane];
            v8f acc = {0.f, 0.f, 0.f, 0.f, 0.f, 0.f, 0.f, 0.f};
            acc = __builtin_amdgcn_wmma_f32_16x16x32_f16(false, a0, false, b0,
                                                         (short)0, acc, false, false);
            acc = __builtin_amdgcn_wmma_f32_16x16x32_f16(false, a1, false, b1,
                                                         (short)0, acc, false, false);
            int col = nt * 16 + (lane & 15);
            int mo  = ((lane >> 4) & 1) * 8;
            float bb = glob_b[col];
#pragma unroll
            for (int i2 = 0; i2 < 8; ++i2)
                sG[wave][(mo + i2) * 32 + col] = fmaxf(acc[i2] + bb, 0.f);
        }
    }

    // agg weights + softmax over samples + weighted sum
    if (lane < 16) {
        float acc = aggw_b[0];
        for (int c = 0; c < 32; ++c) acc += sG[wave][lane * 32 + c] * aggw_w[c];
        sW4[wave][lane] = fmaxf(acc, 0.f);
    }
    if (lane < 4) softmax4(&sW4[wave][lane * 4]);
    for (int i = lane; i < 4 * 32; i += 32) {
        int p = i >> 5, c = i & 31;
        float acc = 0.f;
        for (int sm = 0; sm < 4; ++sm)
            acc += sG[wave][(p * 4 + sm) * 32 + c] * sW4[wave][p * 4 + sm];
        sIMF[wave][i] = acc;
    }
    // fc (32->16) + build x = [vox(8) | img_feat(16)]
    for (int i = lane; i < 4 * 24; i += 32) {
        int p = i / 24, c = i - p * 24;
        float v;
        if (c < 8) v = vox[(size_t)(p0 + p) * 8 + c];
        else {
            int j = c - 8;
            float acc = fc_b[j];
            for (int k = 0; k < 32; ++k) acc += sIMF[wave][p * 32 + k] * fc_w[k * 16 + j];
            v = fmaxf(acc, 0.f);
        }
        sXB[wave][i] = v;
        Xout[(size_t)(p0 + p) * 24 + c] = v;
    }
    // sigma = softplus(x @ sig_w + b)
    if (lane < 4) {
        float acc = sig_b[0];
        for (int c = 0; c < 24; ++c) acc += sXB[wave][lane * 24 + c] * sig_w[c];
        SIGout[p0 + lane] = softplusf(acc);
    }
    // pack x0 A-matrix (fragment order): logical [row][K] = [x(24)|f(23)|0]
    for (int i = lane; i < 2 * 32 * 16; i += 32) {
        int kt = i >> 9, rem = i & 511, l2 = rem >> 4, h = rem & 15;
        int row = l2 & 15;
        int k = kt * 32 + ((h >> 3) << 4) + ((l2 >> 4) & 1) * 8 + (h & 7);
        float v = 0.f;
        if      (k < 24) v = sXB[wave][(row >> 2) * 24 + k];
        else if (k < 47) v = F[row * 23 + (k - 24)];
        A[i] = (_Float16)v;
    }

    // ---- WMMA GEMM #2: [16,47] x [47,24] -> col hidden ----
    {
        const v16h* Af = reinterpret_cast<const v16h*>(A);
        const v16h* Bf = reinterpret_cast<const v16h*>(sCol1W);
        v16h a0 = Af[lane];
        v16h a1 = Af[32 + lane];
#pragma unroll
        for (int nt = 0; nt < 2; ++nt) {
            v16h b0 = Bf[(0 * 2 + nt) * 32 + lane];
            v16h b1 = Bf[(1 * 2 + nt) * 32 + lane];
            v8f acc = {0.f, 0.f, 0.f, 0.f, 0.f, 0.f, 0.f, 0.f};
            acc = __builtin_amdgcn_wmma_f32_16x16x32_f16(false, a0, false, b0,
                                                         (short)0, acc, false, false);
            acc = __builtin_amdgcn_wmma_f32_16x16x32_f16(false, a1, false, b1,
                                                         (short)0, acc, false, false);
            int col = nt * 16 + (lane & 15);
            int mo  = ((lane >> 4) & 1) * 8;
            if (col < 24) {
                float bb = col1_b[col];
#pragma unroll
                for (int i2 = 0; i2 < 8; ++i2)
                    sH[wave][(mo + i2) * 32 + col] = fmaxf(acc[i2] + bb, 0.f);
            }
        }
    }

    // col2 + softmax over samples + radiance
    if (lane < 16) {
        float acc = col2_b[0];
        for (int c = 0; c < 24; ++c) acc += sH[wave][lane * 32 + c] * col2_w[c];
        sW4[wave][lane] = fmaxf(acc, 0.f);
    }
    if (lane < 4) softmax4(&sW4[wave][lane * 4]);
    if (lane < 12) {
        int p = lane / 3, c = lane - p * 3;
        float acc = 0.f;
        for (int sm = 0; sm < 4; ++sm)
            acc += F[(p * 4 + sm) * 23 + 16 + c] * sW4[wave][p * 4 + sm];
        RADout[(size_t)(p0 + p) * 3 + c] = acc;
    }
}

// =====================================================================
// Kernel 2: volume rendering per ray
// =====================================================================
__global__ __launch_bounds__(256)
void k2_volrender(const float* __restrict__ X, const float* __restrict__ SIG,
                  const float* __restrict__ RAD, const float* __restrict__ zvals,
                  float* __restrict__ xaPlane /*[24][HW]*/,
                  float* __restrict__ depthOut /*[HW]*/,
                  float* __restrict__ rgbOut /*[3][HW] -> d_out seg5*/)
{
    int ray = blockIdx.x * blockDim.x + threadIdx.x;
    if (ray >= HWPIX) return;
    float T = 1.f;
    float xa[24];
#pragma unroll
    for (int c = 0; c < 24; ++c) xa[c] = 0.f;
    float rgb[3] = {0.f, 0.f, 0.f};
    float depth = 0.f;
#pragma unroll
    for (int j = 0; j < 8; ++j) {
        size_t pi = (size_t)ray * 8 + j;
        float alpha = 1.f - __expf(-SIG[pi]);
        float w = alpha * T;
        T *= (1.f - alpha + 1e-10f);
        const float* xp = X + pi * 24;
#pragma unroll
        for (int c = 0; c < 24; ++c) xa[c] += w * xp[c];
        const float* rp = RAD + pi * 3;
        rgb[0] += w * rp[0]; rgb[1] += w * rp[1]; rgb[2] += w * rp[2];
        depth += w * (1.f / fmaxf(zvals[pi], 1e-6f));
    }
#pragma unroll
    for (int c = 0; c < 24; ++c) xaPlane[(size_t)c * HWPIX + ray] = xa[c];
    rgbOut[ray] = rgb[0];
    rgbOut[HWPIX + ray] = rgb[1];
    rgbOut[2 * HWPIX + ray] = rgb[2];
    depthOut[ray] = depth;
}

// =====================================================================
// Kernel 3: generic 3x3 conv (SAME, stride 1/2, optional nearest x2 up)
// =====================================================================
__global__ __launch_bounds__(256)
void k_conv3x3(const float* __restrict__ in, const float* __restrict__ w,
               const float* __restrict__ bias, float* __restrict__ out,
               int CIN, int COUT, int HL, int stride, int upsample,
               int doRelu, int interleave)
{
    int HO = HL / stride, WO = HO;
    int total = HO * WO * COUT;
    int idx = blockIdx.x * blockDim.x + threadIdx.x;
    if (idx >= total) return;
    int pix = idx % (HO * WO);
    int o   = idx / (HO * WO);
    int ox = pix % WO, oy = pix / WO;
    int HP = upsample ? (HL >> 1) : HL;
    int padlo = (stride == 1) ? 1 : 0;     // JAX SAME: s=2 -> pad_lo 0, pad_hi 1
    float acc = bias[o];
    for (int ci = 0; ci < CIN; ++ci) {
        const float* wp = w + ((size_t)o * CIN + ci) * 9;
        const float* ip = in + (size_t)ci * HP * HP;
        // keep the next channel plane's lines in flight (global_prefetch_b8)
        if (ci + 1 < CIN) {
            int pyp = upsample ? ((oy * stride) >> 1) : (oy * stride);
            __builtin_prefetch(ip + (size_t)HP * HP + (size_t)pyp * HP, 0, 1);
        }
        for (int ky = 0; ky < 3; ++ky) {
            int iy = oy * stride + ky - padlo;
            if (iy < 0 || iy >= HL) continue;
            int py = upsample ? (iy >> 1) : iy;
            for (int kx = 0; kx < 3; ++kx) {
                int ix = ox * stride + kx - padlo;
                if (ix < 0 || ix >= HL) continue;
                int px = upsample ? (ix >> 1) : ix;
                acc += ip[(size_t)py * HP + px] * wp[ky * 3 + kx];
            }
        }
    }
    if (doRelu) acc = fmaxf(acc, 0.f);
    if (interleave) out[(size_t)pix * COUT + o] = acc;
    else            out[(size_t)o * HO * WO + pix] = acc;
}

// =====================================================================
// Kernel 4: invert ixt (3x3) and ext (4x4) -> ws
// =====================================================================
__global__ void k_inv(const float* __restrict__ ext, const float* __restrict__ ixt,
                      float* __restrict__ out /*[9 invIxt | 16 invExt]*/)
{
    if (threadIdx.x != 0 || blockIdx.x != 0) return;
    float a = ixt[0], b = ixt[1], c = ixt[2];
    float d = ixt[3], e = ixt[4], f = ixt[5];
    float g = ixt[6], h = ixt[7], i = ixt[8];
    float A = e * i - f * h, B = -(d * i - f * g), C = d * h - e * g;
    float inv = 1.f / (a * A + b * B + c * C);
    out[0] = A * inv;            out[1] = (c * h - b * i) * inv;  out[2] = (b * f - c * e) * inv;
    out[3] = B * inv;            out[4] = (a * i - c * g) * inv;  out[5] = (c * d - a * f) * inv;
    out[6] = C * inv;            out[7] = (b * g - a * h) * inv;  out[8] = (a * e - b * d) * inv;
    // 4x4 Gauss-Jordan with partial pivoting
    float M[4][8];
    for (int r = 0; r < 4; ++r)
        for (int cc = 0; cc < 4; ++cc) {
            M[r][cc] = ext[r * 4 + cc];
            M[r][4 + cc] = (r == cc) ? 1.f : 0.f;
        }
    for (int col = 0; col < 4; ++col) {
        int piv = col;
        for (int r = col + 1; r < 4; ++r)
            if (fabsf(M[r][col]) > fabsf(M[piv][col])) piv = r;
        if (piv != col)
            for (int cc = 0; cc < 8; ++cc) {
                float t = M[col][cc]; M[col][cc] = M[piv][cc]; M[piv][cc] = t;
            }
        float p = 1.f / M[col][col];
        for (int cc = 0; cc < 8; ++cc) M[col][cc] *= p;
        for (int r = 0; r < 4; ++r) {
            if (r == col) continue;
            float fm = M[r][col];
            for (int cc = 0; cc < 8; ++cc) M[r][cc] -= fm * M[col][cc];
        }
    }
    for (int r = 0; r < 4; ++r)
        for (int cc = 0; cc < 4; ++cc)
            out[9 + r * 4 + cc] = M[r][4 + cc];
}

// =====================================================================
// Kernel 5: per-pixel heads + unprojection
// =====================================================================
__global__ __launch_bounds__(256)
void k_heads(const float* __restrict__ xf /*[HW][24]*/,
             const float* __restrict__ rgbPlane /*[3][HW]*/,
             const float* __restrict__ depth /*[HW]*/,
             const float* __restrict__ invm /*[25]*/,
             const float* __restrict__ op1_w, const float* __restrict__ op1_b,
             const float* __restrict__ op2_w, const float* __restrict__ op2_b,
             const float* __restrict__ sc1_w, const float* __restrict__ sc1_b,
             const float* __restrict__ sc2_w, const float* __restrict__ sc2_b,
             const float* __restrict__ cg1_w, const float* __restrict__ cg1_b,
             const float* __restrict__ cg2_w, const float* __restrict__ cg2_b,
             float* __restrict__ outW, float* __restrict__ outRot,
             float* __restrict__ outScale, float* __restrict__ outOp,
             float* __restrict__ outCol)
{
    int pix = blockIdx.x * blockDim.x + threadIdx.x;
    if (pix >= HWPIX) return;
    float x[27];
#pragma unroll
    for (int c = 0; c < 24; ++c) x[c] = xf[(size_t)pix * 24 + c];
    x[24] = rgbPlane[pix];
    x[25] = rgbPlane[HWPIX + pix];
    x[26] = rgbPlane[2 * HWPIX + pix];
    float t[24];
    // scale head
    for (int o = 0; o < 24; ++o) {
        float a = sc1_b[o];
        for (int i = 0; i < 24; ++i) a += x[i] * sc1_w[i * 24 + o];
        t[o] = fmaxf(a, 0.f);
    }
    for (int o = 0; o < 3; ++o) {
        float a = sc2_b[o];
        for (int i = 0; i < 24; ++i) a += t[i] * sc2_w[i * 3 + o];
        outScale[(size_t)pix * 3 + o] = softplusf(a);
    }
    // opacity head
    for (int o = 0; o < 24; ++o) {
        float a = op1_b[o];
        for (int i = 0; i < 24; ++i) a += x[i] * op1_w[i * 24 + o];
        t[o] = fmaxf(a, 0.f);
    }
    {
        float a = op2_b[0];
        for (int i = 0; i < 24; ++i) a += t[i] * op2_w[i];
        outOp[pix] = sigmoidf(a);
    }
    // color head (input 27)
    for (int o = 0; o < 24; ++o) {
        float a = cg1_b[o];
        for (int i = 0; i < 27; ++i) a += x[i] * cg1_w[i * 24 + o];
        t[o] = fmaxf(a, 0.f);
    }
    for (int o = 0; o < 3; ++o) {
        float a = cg2_b[o];
        for (int i = 0; i < 24; ++i) a += t[i] * cg2_w[i * 3 + o];
        outCol[(size_t)pix * 3 + o] = sigmoidf(a);
    }
    // rotation = normalize(ones) = 0.5
    outRot[(size_t)pix * 4 + 0] = 0.5f;
    outRot[(size_t)pix * 4 + 1] = 0.5f;
    outRot[(size_t)pix * 4 + 2] = 0.5f;
    outRot[(size_t)pix * 4 + 3] = 0.5f;
    // unprojection
    float dz = depth[pix];
    float fx = (float)(pix % IMGW), fy = (float)(pix / IMGW);
    const float* Ki = invm;
    const float* Ei = invm + 9;
    float cx = (Ki[0] * fx + Ki[1] * fy + Ki[2]) * dz;
    float cy = (Ki[3] * fx + Ki[4] * fy + Ki[5]) * dz;
    float cz = (Ki[6] * fx + Ki[7] * fy + Ki[8]) * dz;
    outW[(size_t)pix * 3 + 0] = Ei[0] * cx + Ei[1] * cy + Ei[2]  * cz + Ei[3];
    outW[(size_t)pix * 3 + 1] = Ei[4] * cx + Ei[5] * cy + Ei[6]  * cz + Ei[7];
    outW[(size_t)pix * 3 + 2] = Ei[8] * cx + Ei[9] * cy + Ei[10] * cz + Ei[11];
}

// =====================================================================
extern "C" void kernel_launch(void* const* d_in, const int* in_sizes, int n_in,
                              void* d_out, int out_size, void* d_ws, size_t ws_size,
                              hipStream_t stream) {
    const float* vox    = (const float*)d_in[0];
    const float* fin    = (const float*)d_in[1];
    const float* zvals  = (const float*)d_in[2];
    const float* ext    = (const float*)d_in[3];
    const float* ixt    = (const float*)d_in[4];
    const float* view_w = (const float*)d_in[5];
    const float* view_b = (const float*)d_in[6];
    const float* glob_w = (const float*)d_in[7];
    const float* glob_b = (const float*)d_in[8];
    const float* aggw_w = (const float*)d_in[9];
    const float* aggw_b = (const float*)d_in[10];
    const float* fc_w   = (const float*)d_in[11];
    const float* fc_b   = (const float*)d_in[12];
    const float* op1_w  = (const float*)d_in[13];
    const float* op1_b  = (const float*)d_in[14];
    const float* op2_w  = (const float*)d_in[15];
    const float* op2_b  = (const float*)d_in[16];
    const float* sc1_w  = (const float*)d_in[17];
    const float* sc1_b  = (const float*)d_in[18];
    const float* sc2_w  = (const float*)d_in[19];
    const float* sc2_b  = (const float*)d_in[20];
    const float* col1_w = (const float*)d_in[21];
    const float* col1_b = (const float*)d_in[22];
    const float* col2_w = (const float*)d_in[23];
    const float* col2_b = (const float*)d_in[24];
    const float* sig_w  = (const float*)d_in[25];
    const float* sig_b  = (const float*)d_in[26];
    const float* cg1_w  = (const float*)d_in[27];
    const float* cg1_b  = (const float*)d_in[28];
    const float* cg2_w  = (const float*)d_in[29];
    const float* cg2_b  = (const float*)d_in[30];
    const float* uc1_w  = (const float*)d_in[31];
    const float* uc1_b  = (const float*)d_in[32];
    const float* uc2_w  = (const float*)d_in[33];
    const float* uc2_b  = (const float*)d_in[34];
    const float* uc3_w  = (const float*)d_in[35];
    const float* uc3_b  = (const float*)d_in[36];
    const float* uc4_w  = (const float*)d_in[37];
    const float* uc4_b  = (const float*)d_in[38];

    float* out = (float*)d_out;
    float* ws  = (float*)d_ws;

    const size_t N  = NPTS;
    const size_t HW = HWPIX;

    // workspace layout
    float* X     = ws;                       // [N][24]
    float* SIG   = X + N * 24;               // [N]
    float* RAD   = SIG + N;                  // [N][3]
    float* xa    = RAD + N * 3;              // [24][HW]
    float* depth = xa + 24 * HW;             // [HW]
    float* invm  = depth + HW;               // [25]
    // UNet buffers reuse the X region (X consumed by kernel 2)
    float* hcat  = X;                        // [32][HW]  (h1 planes 0..15, h3 planes 16..31)
    float* h2    = X + 32 * HW;              // [32][128*128]
    float* xf    = h2 + 32 * HW / 4;         // [HW][24]

    // output segments (reference tuple order)
    float* out_world = out;                  // [HW][3]
    float* out_rot   = out + HW * 3;         // [HW][4]
    float* out_scale = out + HW * 7;         // [HW][3]
    float* out_op    = out + HW * 10;        // [HW]
    float* out_col   = out + HW * 11;        // [HW][3]
    float* out_rgb   = out + HW * 14;        // [3][HW]

    // 1) per-point stage (WMMA GEMMs)
    k1_point<<<dim3((unsigned)(N / 16)), dim3(128), 0, stream>>>(
        vox, fin, view_w, view_b, glob_w, glob_b, aggw_w, aggw_b,
        fc_w, fc_b, col1_w, col1_b, col2_w, col2_b, sig_w, sig_b,
        X, SIG, RAD);

    // 2) volume rendering
    k2_volrender<<<dim3((unsigned)(HW / 256)), dim3(256), 0, stream>>>(
        X, SIG, RAD, zvals, xa, depth, out_rgb);

    // 3) UNet
    k_conv3x3<<<dim3((unsigned)((HW * 16 + 255) / 256)), dim3(256), 0, stream>>>(
        xa, uc1_w, uc1_b, hcat, 24, 16, IMGW, 1, 0, 1, 0);
    k_conv3x3<<<dim3((unsigned)((HW / 4 * 32 + 255) / 256)), dim3(256), 0, stream>>>(
        hcat, uc2_w, uc2_b, h2, 16, 32, IMGW, 2, 0, 1, 0);
    k_conv3x3<<<dim3((unsigned)((HW * 16 + 255) / 256)), dim3(256), 0, stream>>>(
        h2, uc3_w, uc3_b, hcat + 16 * HW, 32, 16, IMGW, 1, 1, 1, 0);
    k_conv3x3<<<dim3((unsigned)((HW * 24 + 255) / 256)), dim3(256), 0, stream>>>(
        hcat, uc4_w, uc4_b, xf, 32, 24, IMGW, 1, 0, 0, 1);

    // 4) matrix inverses
    k_inv<<<dim3(1), dim3(1), 0, stream>>>(ext, ixt, invm);

    // 5) per-pixel heads + unprojection
    k_heads<<<dim3((unsigned)(HW / 256)), dim3(256), 0, stream>>>(
        xf, out_rgb, depth, invm,
        op1_w, op1_b, op2_w, op2_b, sc1_w, sc1_b, sc2_w, sc2_b,
        cg1_w, cg1_b, cg2_w, cg2_b,
        out_world, out_rot, out_scale, out_op, out_col);
}
